// DenseDilatedKnnGraph_13546326851629
// MI455X (gfx1250) — matrix operand order repaired and checked by reading, not compiled, not run on hardware
//
#include <hip/hip_runtime.h>

// Problem constants (from reference setup_inputs)
#define BB  4
#define CC  192
#define NN  4096
#define KNN 9
#define TILES_PER_B (NN / 16)   // 256

typedef __attribute__((ext_vector_type(16))) _Float16 v16h;
typedef __attribute__((ext_vector_type(8)))  _Float16 v8h;
typedef __attribute__((ext_vector_type(8)))  float    v8f;

// ---------------------------------------------------------------------------
// Kernel 1: L2-normalize over channels, transpose [B,C,N] -> [B,N,C] as f16,
// and store per-point squared norm of the normalized vector (~1.0).
// ---------------------------------------------------------------------------
__global__ void normalize_kernel(const float* __restrict__ x,
                                 _Float16* __restrict__ xn,
                                 float* __restrict__ sq)
{
    int t = blockIdx.x * blockDim.x + threadIdx.x;   // t = b*N + n
    if (t >= BB * NN) return;
    int b = t / NN;
    int n = t - b * NN;
    const float* xb = x + (size_t)b * CC * NN + n;   // x[b][c][n], stride N over c

    float s = 0.f;
    for (int c = 0; c < CC; ++c) {
        float v = xb[(size_t)c * NN];
        s += v * v;
    }
    float inv = 1.f / fmaxf(sqrtf(s), 1e-12f);

    _Float16* row = xn + (size_t)t * CC;
    float s2 = 0.f;
    for (int c = 0; c < CC; ++c) {
        float v = xb[(size_t)c * NN] * inv;
        row[c] = (_Float16)v;
        s2 += v * v;
    }
    sq[t] = s2;
}

// ---------------------------------------------------------------------------
// Kernel 2: per-batch min/max of gaze (N=4096) -> gmm[2*b], gmm[2*b+1]
// ---------------------------------------------------------------------------
__global__ void gminmax_kernel(const float* __restrict__ g,
                               float* __restrict__ gmm)
{
    __shared__ float smn[256];
    __shared__ float smx[256];
    int b = blockIdx.x, t = threadIdx.x;
    float mn = 3.4e38f, mx = -3.4e38f;
    for (int i = t; i < NN; i += 256) {
        float v = g[b * NN + i];
        mn = fminf(mn, v);
        mx = fmaxf(mx, v);
    }
    smn[t] = mn; smx[t] = mx;
    __syncthreads();
    for (int s = 128; s > 0; s >>= 1) {
        if (t < s) {
            smn[t] = fminf(smn[t], smn[t + s]);
            smx[t] = fmaxf(smx[t], smx[t + s]);
        }
        __syncthreads();
    }
    if (t == 0) { gmm[2 * b] = smn[0]; gmm[2 * b + 1] = smx[0]; }
}

// ---------------------------------------------------------------------------
// Kernel 3: fused WMMA Gram-matrix + gaze-weighted distance + top-9 selection.
// 2 waves per block, each wave owns one 16-row tile (wave32).
// ---------------------------------------------------------------------------
__launch_bounds__(64)
__global__ void knn_wmma_kernel(const _Float16* __restrict__ xn,
                                const float* __restrict__ sq,
                                const float* __restrict__ g,
                                const float* __restrict__ gmm,
                                int* __restrict__ out)
{
    __shared__ float sdist[2][16 * 17];   // padded to dodge bank conflicts
    __shared__ float sA[2][16];           // row-side squared norms
    __shared__ float sG[2][16];           // row-side gaze values
    __shared__ float sGn[2][16];          // row-side normalized gaze weight

    const int wave = threadIdx.x >> 5;
    const int lane = threadIdx.x & 31;
    const int tile = blockIdx.x * 2 + wave;      // 0 .. B*TILES_PER_B-1
    const int b    = tile >> 8;                  // tile / TILES_PER_B
    const int rowbase = (tile & (TILES_PER_B - 1)) << 4;

    const float gmin = gmm[2 * b];
    const float gmax = gmm[2 * b + 1];
    const float ginv = 1.f / (gmax - gmin);

    if (lane < 16) {
        int r = b * NN + rowbase + lane;
        float s  = sq[r];
        float gv = g[r];
        sA [wave][lane] = s;
        sG [wave][lane] = gv;
        sGn[wave][lane] = (gv - gmin) * ginv;
    }
    __syncthreads();

    // --- Preload A operand: 16 rows x 192 f16, 6 chunks of 16x32 -----------
    // 16-bit A layout (contiguous, matches B): lanes 0-15 row M=lane hold K=0..15
    // per chunk; lanes 16-31 hold K=16..31 of row M=lane-16.
    const int m    = lane & 15;
    const int half = lane >> 4;
    const _Float16* arow = xn + ((size_t)(b * NN + rowbase + m)) * CC;
    const int koffA = half * 16;
    v16h Areg[6];
#pragma unroll
    for (int kc = 0; kc < 6; ++kc) {
        v8h lo = *(const v8h*)(arow + kc * 32 + koffA);
        v8h hi = *(const v8h*)(arow + kc * 32 + 8 + koffA);
        Areg[kc] = __builtin_shufflevector(lo, hi,
            0, 1, 2, 3, 4, 5, 6, 7, 8, 9, 10, 11, 12, 13, 14, 15);
    }

    // register-resident top-9 (ascending distance), fully unrolled
    float bd[KNN]; int bi[KNN];
#pragma unroll
    for (int k = 0; k < KNN; ++k) { bd[k] = 3.4e38f; bi[k] = 0; }

    // 16-bit B layout (32x16): lanes 0-15 col N=lane hold K=0..15,
    // lanes 16-31 hold K=16..31 -> one contiguous 32B read per chunk.
    const int koffB  = half * 16;
    const int rowoff = half * 8;

    for (int ct = 0; ct < TILES_PER_B; ++ct) {
        const int colbase = ct << 4;
        const _Float16* bcol =
            xn + ((size_t)(b * NN + colbase + m)) * CC;
        if (ct + 1 < TILES_PER_B)
            __builtin_prefetch(bcol + 16 * CC, 0, 1);   // global_prefetch_b8

        v8f acc = {};
#pragma unroll
        for (int kc = 0; kc < 6; ++kc) {
            v16h bm = *(const v16h*)(bcol + kc * 32 + koffB);
            acc = __builtin_amdgcn_wmma_f32_16x16x32_f16(
                      false, Areg[kc], false, bm,
                      (short)0, acc, false, false);
        }

        // Epilogue: dist = |a|^2 + |b|^2 - 2 a.b + (ga-gb)^2 * gnorm_a
        const float sqB = sq[b * NN + colbase + m];
        const float gB  = g [b * NN + colbase + m];
#pragma unroll
        for (int i = 0; i < 8; ++i) {
            int   r  = i + rowoff;                      // C-tile row mapping
            float d  = sA[wave][r] + sqB - 2.f * acc[i];
            float gd = sG[wave][r] - gB;
            d += gd * gd * sGn[wave][r];
            sdist[wave][r * 17 + m] = d;
        }
        __syncthreads();

        if (lane < 16) {
#pragma unroll 4
            for (int j = 0; j < 16; ++j) {
                float d = sdist[wave][lane * 17 + j];
                if (d < bd[KNN - 1]) {
                    bd[KNN - 1] = d;
                    bi[KNN - 1] = colbase + j;
#pragma unroll
                    for (int s = KNN - 1; s > 0; --s) {
                        if (bd[s] < bd[s - 1]) {
                            float tf = bd[s]; bd[s] = bd[s - 1]; bd[s - 1] = tf;
                            int   ti = bi[s]; bi[s] = bi[s - 1]; bi[s - 1] = ti;
                        }
                    }
                }
            }
        }
        __syncthreads();
    }

    // edge_index[0,b,n,k] = nn_idx, edge_index[1,b,n,k] = n  (int32)
    if (lane < 16) {
        int n = rowbase + lane;
        int* o0 = out + ((size_t)b * NN + n) * KNN;
        int* o1 = out + ((size_t)(BB + b) * NN + n) * KNN;
#pragma unroll
        for (int k = 0; k < KNN; ++k) { o0[k] = bi[k]; o1[k] = n; }
    }
}

// ---------------------------------------------------------------------------
extern "C" void kernel_launch(void* const* d_in, const int* in_sizes, int n_in,
                              void* d_out, int out_size, void* d_ws, size_t ws_size,
                              hipStream_t stream)
{
    const float* x    = (const float*)d_in[0];   // [B,C,N,1] f32
    const float* gaze = (const float*)d_in[1];   // [B,1,N,1] f32 -> g[b*N+n]

    // workspace layout
    _Float16* xn  = (_Float16*)d_ws;                                  // B*N*C f16
    float*    sq  = (float*)((char*)d_ws + (size_t)BB * NN * CC * 2); // B*N f32
    float*    gmm = sq + (size_t)BB * NN;                             // 2*B f32

    int* out = (int*)d_out;                      // [2,B,N,9] int32

    normalize_kernel<<<(BB * NN) / 256, 256, 0, stream>>>(x, xn, sq);
    gminmax_kernel<<<BB, 256, 0, stream>>>(gaze, gmm);
    knn_wmma_kernel<<<(BB * NN) / 32, 64, 0, stream>>>(xn, sq, gaze, gmm, out);
}
